// G2AAgent_6047313952790
// MI455X (gfx1250) — matrix operand order repaired and checked by reading, not compiled, not run on hardware
//
#include <hip/hip_runtime.h>
#include <cstdint>
#include <cstddef>

// ---------------- problem constants ----------------
#define A_N     16
#define H_N     128
#define IN_N    128
#define NA_N    32
#define NACT_N  14
#define BZ_N    16384
#define BENV_N  1024
#define H3_N    384     // 3*H

typedef _Float16 v16h __attribute__((ext_vector_type(16)));
typedef _Float16 v8h  __attribute__((ext_vector_type(8)));
typedef float    v8f  __attribute__((ext_vector_type(8)));

// ---------------- WMMA helpers ----------------
static __device__ __forceinline__ v8f wmma_f16(v16h a, v16h b, v8f c) {
  // D(16x16,f32) = A(16x32,f16) * B(32x16,f16) + C
  return __builtin_amdgcn_wmma_f32_16x16x32_f16(false, a, false, b, (short)0, c, false, false);
}

static __device__ __forceinline__ v8f bcast8(float x) {
  v8f r;
#pragma unroll
  for (int i = 0; i < 8; ++i) r[i] = x;
  return r;
}

// A-fragment (16x32 f16). rowPtr = &tile[row][0] for this lane's row (row = lane&15).
// Lanes 0-15 hold K {k0..k0+7, k0+16..k0+23}; lanes 16-31 hold {k0+8..k0+15, k0+24..k0+31}.
static __device__ __forceinline__ v16h load_a_row(const _Float16* rowPtr, int k0, int hi) {
  const _Float16* p = rowPtr + k0 + (hi << 3);
  v8h lo = *(const v8h*)(p);
  v8h hb = *(const v8h*)(p + 16);
  v16h r;
#pragma unroll
  for (int i = 0; i < 8; ++i) { r[i] = lo[i]; r[i + 8] = hb[i]; }
  return r;
}

// B-fragment (32x16 f16) from row-major weight W[N][K]: lane l holds column n0+l,
// K {k0..k0+7, k0+16..k0+23} (lanes 0-15) or {k0+8..k0+15, k0+24..k0+31} (lanes 16-31).
static __device__ __forceinline__ v16h load_b(const _Float16* W, int ldk, int n0, int k0,
                                              int l, int hi) {
  const _Float16* p = W + (size_t)(n0 + l) * ldk + k0 + (hi << 3);
  v8h lo = *(const v8h*)(p);
  v8h hb = *(const v8h*)(p + 16);
  v16h r;
#pragma unroll
  for (int i = 0; i < 8; ++i) { r[i] = lo[i]; r[i + 8] = hb[i]; }
  return r;
}

static __device__ __forceinline__ float sigf(float x) { return 1.0f / (1.0f + __expf(-x)); }

// CDNA5 async global->LDS copy of 16 bytes (ASYNCcnt-tracked, bypasses VGPRs).
// ldsByteOff is the byte offset within the block's dynamic-LDS allocation.
static __device__ __forceinline__ void async_g2l_b128(unsigned int ldsByteOff,
                                                      const void* gptr) {
  unsigned long long ga = (unsigned long long)(uintptr_t)gptr;
  asm volatile("global_load_async_to_lds_b128 %0, %1, off"
               :: "v"(ldsByteOff), "v"(ga)
               : "memory");
}
static __device__ __forceinline__ void wait_asynccnt0() {
  asm volatile("s_wait_asynccnt 0" ::: "memory");
}

// ---------------- kernel 0: f32 -> f16 weight conversion ----------------
__global__ void k_cvt(const float* __restrict__ s, _Float16* __restrict__ d, int n) {
  int i = blockIdx.x * 256 + threadIdx.x;
  if (i < n) d[i] = (_Float16)s[i];
}

// Build zero-padded [16][128] f16 tiles of Wc[:, 0:128] (fwd) and Wc[:, 128:256] (bwd).
__global__ void k_wcpad(const float* __restrict__ Wc, _Float16* __restrict__ dF,
                        _Float16* __restrict__ dB) {
  int i = blockIdx.x * 256 + threadIdx.x;
  if (i < 2048) {
    int row = i >> 7, cc = i & 127;
    dF[i] = (row < 2) ? (_Float16)Wc[row * 256 + cc] : (_Float16)0.0f;
    dB[i] = (row < 2) ? (_Float16)Wc[row * 256 + 128 + cc] : (_Float16)0.0f;
  }
}

// ---------------- kernel 1: x1 = relu(inputs@W1^T+b1); h_rnn = GRUcell(x1, h0) ----------------
__global__ void __launch_bounds__(128)
k1_cell(const float* __restrict__ inputs, const float* __restrict__ hidden,
        const _Float16* __restrict__ W1h, const float* __restrict__ b1,
        const _Float16* __restrict__ WihC, const _Float16* __restrict__ WhhC,
        const float* __restrict__ bihc, const float* __restrict__ bhhc,
        float* __restrict__ outH, _Float16* __restrict__ hF16) {
  __shared__ _Float16 sXin[4][16 * 128];
  __shared__ _Float16 sX1 [4][16 * 128];
  __shared__ _Float16 sHp [4][16 * 128];
  const int tid = threadIdx.x, lane = tid & 31, w = tid >> 5;
  const int l = lane & 15, hi = lane >> 4;
  const int rb = (blockIdx.x * 4 + w) * 16;   // first global row of this wave's 16-row tile

  for (int i = lane; i < 2048; i += 32) {
    sXin[w][i] = (_Float16)inputs[(size_t)rb * 128 + i];
    sHp [w][i] = (_Float16)hidden[(size_t)rb * 128 + i];
  }
  v16h xinF[4], hpF[4];
#pragma unroll
  for (int kt = 0; kt < 4; ++kt) {
    xinF[kt] = load_a_row(&sXin[w][l * 128], kt * 32, hi);
    hpF [kt] = load_a_row(&sHp [w][l * 128], kt * 32, hi);
  }
  // GEMM1 + relu -> sX1 (f16)
#pragma unroll
  for (int nt = 0; nt < 8; ++nt) {
    v8f acc = bcast8(b1[nt * 16 + l]);
#pragma unroll
    for (int kt = 0; kt < 4; ++kt)
      acc = wmma_f16(xinF[kt], load_b(W1h, 128, nt * 16, kt * 32, l, hi), acc);
#pragma unroll
    for (int j = 0; j < 8; ++j) {
      const int row = j + hi * 8, col = nt * 16 + l;
      float v = acc[j];
      sX1[w][row * 128 + col] = (_Float16)(v > 0.0f ? v : 0.0f);
    }
  }
  v16h x1F[4];
#pragma unroll
  for (int kt = 0; kt < 4; ++kt) x1F[kt] = load_a_row(&sX1[w][l * 128], kt * 32, hi);

  float brz[16], bin[8], bhn[8];
#pragma unroll
  for (int nt = 0; nt < 16; ++nt) brz[nt] = bihc[nt * 16 + l] + bhhc[nt * 16 + l];
#pragma unroll
  for (int ct = 0; ct < 8; ++ct) {
    bin[ct] = bihc[256 + ct * 16 + l];
    bhn[ct] = bhhc[256 + ct * 16 + l];
  }
  v8f rz[16];
#pragma unroll
  for (int nt = 0; nt < 16; ++nt) {
    v8f acc = bcast8(brz[nt]);
#pragma unroll
    for (int kt = 0; kt < 4; ++kt)
      acc = wmma_f16(x1F[kt], load_b(WihC, 128, nt * 16, kt * 32, l, hi), acc);
#pragma unroll
    for (int kt = 0; kt < 4; ++kt)
      acc = wmma_f16(hpF[kt], load_b(WhhC, 128, nt * 16, kt * 32, l, hi), acc);
    rz[nt] = acc;
  }
#pragma unroll
  for (int ct = 0; ct < 8; ++ct) {
    v8f gi = bcast8(bin[ct]);
    v8f gh = bcast8(bhn[ct]);
#pragma unroll
    for (int kt = 0; kt < 4; ++kt)
      gi = wmma_f16(x1F[kt], load_b(WihC, 128, 256 + ct * 16, kt * 32, l, hi), gi);
#pragma unroll
    for (int kt = 0; kt < 4; ++kt)
      gh = wmma_f16(hpF[kt], load_b(WhhC, 128, 256 + ct * 16, kt * 32, l, hi), gh);
#pragma unroll
    for (int j = 0; j < 8; ++j) {
      const int row = j + hi * 8, col = ct * 16 + l;
      float r = sigf(rz[ct][j]);
      float z = sigf(rz[ct + 8][j]);
      float n = tanhf(gi[j] + r * gh[j]);
      float hp = hidden[(size_t)(rb + row) * 128 + col];
      float hv = (1.0f - z) * n + z * hp;
      outH[(size_t)(rb + row) * 128 + col] = hv;
      hF16[(size_t)(rb + row) * 128 + col] = (_Float16)hv;
    }
  }
}

// ---------------- kernel 2: 15-step pair-GRU scan (one wave = one env of 16 agents) ----------------
// Emits partial Gumbel logits lp{0,1}[bz*15+m] = y_dir[m] . Wc[c, dirOff:dirOff+128]
// via a zero-padded (N=2->16) WMMA against wcPad, reusing the step's hidden A-fragments.
__global__ void __launch_bounds__(256)
k2_gru(const _Float16* __restrict__ hF16,
       const _Float16* __restrict__ Wih,    // [384][256] f16
       const _Float16* __restrict__ Whh,    // [384][128] f16
       const float* __restrict__ bih, const float* __restrict__ bhh,
       const _Float16* __restrict__ wcPad,  // [16][128] f16, rows 0/1 = Wc classes
       float* __restrict__ lp0, float* __restrict__ lp1, int backward) {
  extern __shared__ char smem[];
  _Float16* sWhh = (_Float16*)smem;                          // 384*128 f16 = 96KB @ offset 0
  const int tid = threadIdx.x, lane = tid & 31, w = tid >> 5;
  const int l = lane & 15, hi = lane >> 4;
  _Float16* h3t = (_Float16*)(smem + 98304) + (size_t)w * 4096;  // 16x128 f16
  _Float16* hc  = h3t + 2048;                                    // 16x128 f16

  const int e = blockIdx.x * 8 + w;
  const int rb = e * 16;

  // stage Whh (block-cooperative) and h3 env tile (per wave) with async global->LDS DMA
  for (int i = tid; i < 6144; i += 256)
    async_g2l_b128((unsigned)(i * 16), (const char*)Whh + (size_t)i * 16);
  {
    const char* src = (const char*)(hF16 + (size_t)rb * 128);
    unsigned base = 98304u + (unsigned)w * 8192u;
    for (int i = lane; i < 256; i += 32)
      async_g2l_b128(base + (unsigned)i * 16, src + (size_t)i * 16);
  }
  // zero current hidden state
  {
    v8h zz;
#pragma unroll
    for (int i = 0; i < 8; ++i) zz[i] = (_Float16)0.0f;
    for (int i = lane; i < 256; i += 32) ((v8h*)hc)[i] = zz;
  }
  wait_asynccnt0();
  __syncthreads();

  float brz[16], bin[8], bhn[8];
#pragma unroll
  for (int nt = 0; nt < 16; ++nt) brz[nt] = bih[nt * 16 + l] + bhh[nt * 16 + l];
#pragma unroll
  for (int ct = 0; ct < 8; ++ct) {
    bin[ct] = bih[256 + ct * 16 + l];
    bhn[ct] = bhh[256 + ct * 16 + l];
  }
  // self half of x (h3[a]) is step-invariant: keep A-fragments in VGPRs
  v16h selfA[4];
#pragma unroll
  for (int kt = 0; kt < 4; ++kt) selfA[kt] = load_a_row(h3t + l * 128, kt * 32, hi);

  v16h hA[4];
  // logit projection of the current hidden state (position mPos) via padded WMMA
  auto do_logit = [&](int mPos) {
    v8f acc = bcast8(0.0f);
#pragma unroll
    for (int kt = 0; kt < 4; ++kt)
      acc = wmma_f16(hA[kt], load_b(wcPad, 128, 0, kt * 32, l, hi), acc);
    if (l < 2) {
      float* lp = l ? lp1 : lp0;
#pragma unroll
      for (int j = 0; j < 8; ++j)
        lp[(size_t)(rb + j + hi * 8) * 15 + mPos] = acc[j];
    }
  };

#pragma unroll 1
  for (int t = 0; t < 15; ++t) {
    const int m = backward ? (14 - t) : t;          // sequence position
    const int nb = (m < l) ? m : (m + 1);           // IDX[a=l][m]
    v16h nbA[4];
#pragma unroll
    for (int kt = 0; kt < 4; ++kt) {
      nbA[kt] = load_a_row(h3t + nb * 128, kt * 32, hi);
      hA[kt]  = load_a_row(hc + l * 128,  kt * 32, hi);
    }
    // hA currently holds y[t-1]: emit its logits (positions t-1 fwd / 15-t bwd)
    if (t > 0) do_logit(backward ? (15 - t) : (t - 1));

    // r,z gates: g = x@Wih^T + h@Whh^T + (bih+bhh), N-tiles 0..15 (N 0..255)
    v8f rz[16];
#pragma unroll
    for (int nt = 0; nt < 16; ++nt) {
      v8f acc = bcast8(brz[nt]);
#pragma unroll
      for (int kt = 0; kt < 4; ++kt)
        acc = wmma_f16(selfA[kt], load_b(Wih, 256, nt * 16, kt * 32, l, hi), acc);
#pragma unroll
      for (int kt = 0; kt < 4; ++kt)
        acc = wmma_f16(nbA[kt], load_b(Wih, 256, nt * 16, 128 + kt * 32, l, hi), acc);
#pragma unroll
      for (int kt = 0; kt < 4; ++kt)
        acc = wmma_f16(hA[kt], load_b(sWhh, 128, nt * 16, kt * 32, l, hi), acc);
      rz[nt] = acc;
    }
    // n gate (needs gi_n and gh_n separately) + elementwise GRU update
#pragma unroll
    for (int ct = 0; ct < 8; ++ct) {
      v8f gi = bcast8(bin[ct]);
      v8f gh = bcast8(bhn[ct]);
#pragma unroll
      for (int kt = 0; kt < 4; ++kt)
        gi = wmma_f16(selfA[kt], load_b(Wih, 256, 256 + ct * 16, kt * 32, l, hi), gi);
#pragma unroll
      for (int kt = 0; kt < 4; ++kt)
        gi = wmma_f16(nbA[kt], load_b(Wih, 256, 256 + ct * 16, 128 + kt * 32, l, hi), gi);
#pragma unroll
      for (int kt = 0; kt < 4; ++kt)
        gh = wmma_f16(hA[kt], load_b(sWhh, 128, 256 + ct * 16, kt * 32, l, hi), gh);
#pragma unroll
      for (int j = 0; j < 8; ++j) {
        const int row = j + hi * 8, col = ct * 16 + l;
        float r = sigf(rz[ct][j]);
        float z = sigf(rz[ct + 8][j]);
        float n = tanhf(gi[j] + r * gh[j]);
        float hp = (float)hc[row * 128 + col];
        float hv = (1.0f - z) * n + z * hp;
        hc[row * 128 + col] = (_Float16)hv;
      }
    }
  }
  // epilogue: logits of the final hidden state (position 14 fwd / 0 bwd)
#pragma unroll
  for (int kt = 0; kt < 4; ++kt) hA[kt] = load_a_row(hc + l * 128, kt * 32, hi);
  do_logit(backward ? 0 : 14);
}

// ---------------- kernel 4: gumbel hard-w, q/k attention, aggregation, output head ----------------
__global__ void __launch_bounds__(128)
k4_attn(const _Float16* __restrict__ hF16, const float* __restrict__ gum,
        const float* __restrict__ lf0, const float* __restrict__ lf1,
        const float* __restrict__ lb0, const float* __restrict__ lb1,
        const float* __restrict__ bc,
        const _Float16* __restrict__ Wqh, const _Float16* __restrict__ Wkh,
        const _Float16* __restrict__ W2h, const float* __restrict__ b2,
        float* __restrict__ outO) {
  extern __shared__ char smem[];
  _Float16* w2p = (_Float16*)smem;  // [16][256] f16, rows 14/15 zero-padded
  const int tid = threadIdx.x, lane = tid & 31, w = tid >> 5;
  const int l = lane & 15, hi = lane >> 4;
  for (int i = tid; i < 4096; i += 128) {
    int n = i >> 8, cc = i & 255;
    w2p[i] = (n < NACT_N) ? W2h[n * 256 + cc] : (_Float16)0.0f;
  }
  __syncthreads();

  char* pw = smem + 8192 + (size_t)w * 16384;
  _Float16* xcat = (_Float16*)pw;          // [16][256]: cols 0..127 = h3, 128..255 = agg
  float* qs = (float*)(pw + 8192);         // [16][32]
  float* ks = qs + 512;                    // [16][32]
  float* hw = ks + 512;                    // [16][15]
  float* sc = hw + 240;                    // [16][15]
  float* ww = sc + 240;                    // [16][15]
  const int e = blockIdx.x * 4 + w, rb = e * 16;

  for (int i = lane; i < 2048; i += 32) {
    int row = i >> 7, cc = i & 127;
    xcat[row * 256 + cc] = hF16[(size_t)(rb + row) * 128 + cc];
  }
  // hard gumbel weights
  for (int i = lane; i < 240; i += 32) {
    int a = i / 15, m = i % 15;
    size_t gr = (size_t)(rb + a) * 15 + m;
    float u0 = gum[gr * 2 + 0], u1 = gum[gr * 2 + 1];
    float g0 = -__logf(-__logf(u0 + 1e-10f) + 1e-10f);
    float g1 = -__logf(-__logf(u1 + 1e-10f) + 1e-10f);
    float l0 = lf0[gr] + lb0[gr] + bc[0] + g0;
    float l1 = lf1[gr] + lb1[gr] + bc[1] + g1;
    hw[a * 15 + m] = sigf((l1 - l0) * 2.0f);  // /TAU, TAU=0.5
  }
  // q, k via WMMA
  v16h hF[4];
#pragma unroll
  for (int kt = 0; kt < 4; ++kt) hF[kt] = load_a_row(xcat + l * 256, kt * 32, hi);
#pragma unroll
  for (int nt = 0; nt < 2; ++nt) {
    v8f aq = bcast8(0.0f), ak = bcast8(0.0f);
#pragma unroll
    for (int kt = 0; kt < 4; ++kt) {
      aq = wmma_f16(hF[kt], load_b(Wqh, 128, nt * 16, kt * 32, l, hi), aq);
      ak = wmma_f16(hF[kt], load_b(Wkh, 128, nt * 16, kt * 32, l, hi), ak);
    }
#pragma unroll
    for (int j = 0; j < 8; ++j) {
      int row = j + hi * 8, col = nt * 16 + l;
      qs[row * 32 + col] = aq[j];
      ks[row * 32 + col] = ak[j];
    }
  }
  // scores
  for (int i = lane; i < 240; i += 32) {
    int a = i / 15, m = i % 15;
    int nbi = (m < a) ? m : (m + 1);
    float d = 0.0f;
#pragma unroll 8
    for (int t2 = 0; t2 < 32; ++t2) d += qs[a * 32 + t2] * ks[nbi * 32 + t2];
    sc[a * 15 + m] = hw[a * 15 + m] * d * 0.17677669529663687f;  // 1/sqrt(NA)
  }
  // per-agent softmax over 15 neighbors; w = hard_w * soft_w
  if (lane < 16) {
    int a = lane;
    float mx = -3.0e38f;
    for (int m = 0; m < 15; ++m) mx = fmaxf(mx, sc[a * 15 + m]);
    float s = 0.0f;
    for (int m = 0; m < 15; ++m) s += __expf(sc[a * 15 + m] - mx);
    float inv = 1.0f / s;
    for (int m = 0; m < 15; ++m)
      ww[a * 15 + m] = hw[a * 15 + m] * __expf(sc[a * 15 + m] - mx) * inv;
  }
  // aggregation -> xcat cols 128..255
  for (int i = lane; i < 2048; i += 32) {
    int a = i >> 7, cc = i & 127;
    float s = 0.0f;
#pragma unroll
    for (int m = 0; m < 15; ++m) {
      int nbi = (m < a) ? m : (m + 1);
      s += ww[a * 15 + m] * (float)xcat[nbi * 256 + cc];
    }
    xcat[a * 256 + 128 + cc] = (_Float16)s;
  }
  // out = [h_rnn, agg] @ W2^T + b2  (N=14 padded to 16)
  {
    float bj = (l < NACT_N) ? b2[l] : 0.0f;
    v8f acc = bcast8(bj);
#pragma unroll
    for (int kt = 0; kt < 8; ++kt)
      acc = wmma_f16(load_a_row(xcat + l * 256, kt * 32, hi),
                     load_b(w2p, 256, 0, kt * 32, l, hi), acc);
#pragma unroll
    for (int j = 0; j < 8; ++j) {
      int row = j + hi * 8;
      if (l < NACT_N) outO[(size_t)(rb + row) * NACT_N + l] = acc[j];
    }
  }
}

// ---------------- host launcher ----------------
extern "C" void kernel_launch(void* const* d_in, const int* in_sizes, int n_in,
                              void* d_out, int out_size, void* d_ws, size_t ws_size,
                              hipStream_t stream) {
  (void)in_sizes; (void)n_in; (void)out_size; (void)ws_size;
  const float* inputs = (const float*)d_in[0];
  const float* hidden = (const float*)d_in[1];
  const float* gum    = (const float*)d_in[2];
  const float* W1     = (const float*)d_in[3];
  const float* b1     = (const float*)d_in[4];
  const float* Wih_c  = (const float*)d_in[5];
  const float* Whh_c  = (const float*)d_in[6];
  const float* bih_c  = (const float*)d_in[7];
  const float* bhh_c  = (const float*)d_in[8];
  const float* Wih_f  = (const float*)d_in[9];
  const float* Whh_f  = (const float*)d_in[10];
  const float* bih_f  = (const float*)d_in[11];
  const float* bhh_f  = (const float*)d_in[12];
  const float* Wih_b  = (const float*)d_in[13];
  const float* Whh_b  = (const float*)d_in[14];
  const float* bih_b  = (const float*)d_in[15];
  const float* bhh_b  = (const float*)d_in[16];
  const float* Wc     = (const float*)d_in[17];
  const float* bc     = (const float*)d_in[18];
  const float* Wq     = (const float*)d_in[19];
  const float* Wk     = (const float*)d_in[20];
  const float* W2     = (const float*)d_in[21];
  const float* b2     = (const float*)d_in[22];

  char* ws = (char*)d_ws;
  size_t o = 0;
  auto carve = [&](size_t bytes) -> char* {
    char* p = ws + o;
    o += (bytes + 255) & ~(size_t)255;
    return p;
  };
  _Float16* W1h  = (_Float16*)carve(16384 * 2);
  _Float16* WihC = (_Float16*)carve(49152 * 2);
  _Float16* WhhC = (_Float16*)carve(49152 * 2);
  _Float16* WihF = (_Float16*)carve(98304 * 2);
  _Float16* WhhF = (_Float16*)carve(49152 * 2);
  _Float16* WihB = (_Float16*)carve(98304 * 2);
  _Float16* WhhB = (_Float16*)carve(49152 * 2);
  _Float16* Wqh  = (_Float16*)carve(4096 * 2);
  _Float16* Wkh  = (_Float16*)carve(4096 * 2);
  _Float16* W2h  = (_Float16*)carve(3584 * 2);
  _Float16* wcF  = (_Float16*)carve(2048 * 2);   // padded Wc[:, 0:128]
  _Float16* wcB  = (_Float16*)carve(2048 * 2);   // padded Wc[:, 128:256]
  _Float16* hF16 = (_Float16*)carve((size_t)BZ_N * H_N * 2);
  float* lf0 = (float*)carve((size_t)BZ_N * 15 * 4);
  float* lf1 = (float*)carve((size_t)BZ_N * 15 * 4);
  float* lb0 = (float*)carve((size_t)BZ_N * 15 * 4);
  float* lb1 = (float*)carve((size_t)BZ_N * 15 * 4);

  auto cvt = [&](const float* s, _Float16* d, int n) {
    k_cvt<<<(n + 255) / 256, 256, 0, stream>>>(s, d, n);
  };
  cvt(W1, W1h, 16384);
  cvt(Wih_c, WihC, 49152);
  cvt(Whh_c, WhhC, 49152);
  cvt(Wih_f, WihF, 98304);
  cvt(Whh_f, WhhF, 49152);
  cvt(Wih_b, WihB, 98304);
  cvt(Whh_b, WhhB, 49152);
  cvt(Wq, Wqh, 4096);
  cvt(Wk, Wkh, 4096);
  cvt(W2, W2h, 3584);
  k_wcpad<<<8, 256, 0, stream>>>(Wc, wcF, wcB);

  float* outO = (float*)d_out;
  float* outH = outO + (size_t)BZ_N * NACT_N;

  // kernel 1: 256 blocks x 128 threads (4 waves x 16 rows)
  k1_cell<<<256, 128, 0, stream>>>(inputs, hidden, W1h, b1, WihC, WhhC, bih_c, bhh_c,
                                   outH, hF16);

  // kernel 2: 128 blocks x 256 threads (8 waves = 8 envs), 160KB dynamic LDS
  (void)hipFuncSetAttribute(reinterpret_cast<const void*>(k2_gru),
                            hipFuncAttributeMaxDynamicSharedMemorySize, 163840);
  k2_gru<<<128, 256, 163840, stream>>>(hF16, WihF, WhhF, bih_f, bhh_f, wcF, lf0, lf1, 0);
  k2_gru<<<128, 256, 163840, stream>>>(hF16, WihB, WhhB, bih_b, bhh_b, wcB, lb0, lb1, 1);

  // kernel 4: 256 blocks x 128 threads (4 waves = 4 envs), 72KB dynamic LDS
  (void)hipFuncSetAttribute(reinterpret_cast<const void*>(k4_attn),
                            hipFuncAttributeMaxDynamicSharedMemorySize, 73728);
  k4_attn<<<256, 128, 73728, stream>>>(hF16, gum, lf0, lf1, lb0, lb1, bc, Wqh, Wkh,
                                       W2h, b2, outO);
}
